// DualAttentionBlock_38895223833255
// MI455X (gfx1250) — compile-verified, matrix operand order checked by reading, not run on hardware
//
#include <hip/hip_runtime.h>
#include <hip/hip_bf16.h>
#include <stdint.h>

// ---------------------------------------------------------------------------
// DANet-style dual attention block for MI455X (gfx1250, wave32, WMMA bf16).
// All matmuls use v_wmma_f32_16x16x32_bf16 with fp32 accumulation.
// Position attention is flash-style (online softmax), never materializing the
// 4096x4096 attention matrix.  bf16 conversion uses the native gfx1250 cvt
// path; next-tile global_prefetch_b8 hides HBM latency under softmax math.
// ---------------------------------------------------------------------------

typedef __attribute__((ext_vector_type(16))) __bf16       v16bf;
typedef __attribute__((ext_vector_type(8)))  float        v8f;
typedef __attribute__((ext_vector_type(8)))  unsigned int v8u;
typedef __attribute__((ext_vector_type(4)))  unsigned int v4u;

#define BATCH 4
#define CCH   256
#define NPIX  4096   // 64*64
#define DQK   32     // C/8

// native f32 -> bf16 (RNE) -- lowers to v_cvt_*bf16* on gfx1250
__device__ __forceinline__ unsigned short f2bf(float f) {
    return __builtin_bit_cast(unsigned short, (__bf16)f);
}
__device__ __forceinline__ unsigned int packbf2(float a, float b) {
    return (unsigned int)f2bf(a) | ((unsigned int)f2bf(b) << 16);
}
// K offset of bf16 pair held in fragment VGPR i for lane-half h (ISA 7.12.2).
__device__ __forceinline__ int kofs(int i, int h) {
    return 2 * (i & 3) + 8 * h + 16 * (i >> 2);
}
__device__ __forceinline__ v8f wmma_bf16(v8u a, v8u b, v8f c) {
    return __builtin_amdgcn_wmma_f32_16x16x32_bf16(
        false, __builtin_bit_cast(v16bf, a),
        false, __builtin_bit_cast(v16bf, b),
        (short)0, c, false, false);
}

// ---------------------------------------------------------------------------
// 1) fp32 x -> bf16 copies: xb [B,C,N] (natural) and xT [B,N,C] (transposed)
// ---------------------------------------------------------------------------
__global__ __launch_bounds__(256) void k_cvt_x(const float* __restrict__ x,
                                               unsigned short* __restrict__ xb,
                                               unsigned short* __restrict__ xT) {
    int idx = blockIdx.x * 256 + threadIdx.x;     // < 4*256*4096 = 2^22
    int b   = idx >> 20;                          // 256*4096 = 2^20
    int rem = idx & ((1 << 20) - 1);
    int c   = rem >> 12;
    int n   = rem & (NPIX - 1);
    unsigned short v = f2bf(x[idx]);
    xb[idx] = v;
    xT[((size_t)b * NPIX + n) * CCH + c] = v;
}

// ---------------------------------------------------------------------------
// 2) fusion conv weights [256][512][3][3] fp32 -> wfr [9][256][512] bf16
// ---------------------------------------------------------------------------
__global__ __launch_bounds__(256) void k_cvt_w(const float* __restrict__ wf,
                                               unsigned short* __restrict__ wfr) {
    int idx = blockIdx.x * 256 + threadIdx.x;
    if (idx >= 256 * 512 * 9) return;
    int co = idx / (512 * 9);
    int r2 = idx - co * 512 * 9;
    int ci = r2 / 9;
    int r  = r2 - ci * 9;
    wfr[((size_t)r * 256 + co) * 512 + ci] = f2bf(wf[idx]);
}

// ---------------------------------------------------------------------------
// 3) QKV projection (1x1 convs) via WMMA.
//    out = W[Cout,256] @ x[256,N] + bias;  q,k stored transposed [n][32] bf16,
//    v stored natural [c][n] bf16 (layouts chosen for contiguous frag loads).
//    One wave per 16x16 output tile; mt 0..19 -> {q0,q1,k0,k1,v0..v15}.
// ---------------------------------------------------------------------------
__global__ __launch_bounds__(32) void k_proj(
    const float* __restrict__ wq, const float* __restrict__ bq,
    const float* __restrict__ wk, const float* __restrict__ bk,
    const float* __restrict__ wv, const float* __restrict__ bv,
    const unsigned short* __restrict__ xT,
    unsigned short* __restrict__ qT, unsigned short* __restrict__ kT,
    unsigned short* __restrict__ vb) {
    int wid = blockIdx.x;
    int b   = wid / (20 * 256);
    int r   = wid - b * 20 * 256;
    int mt  = r >> 8;
    int n0  = (r & 255) << 4;
    int L = threadIdx.x, h = L >> 4, m = L & 15;

    const float* W; const float* bias; int row0; int mode;
    if (mt < 2)      { W = wq; bias = bq; row0 = mt * 16;        mode = 0; }
    else if (mt < 4) { W = wk; bias = bk; row0 = (mt - 2) * 16;  mode = 1; }
    else             { W = wv; bias = bv; row0 = (mt - 4) * 16;  mode = 2; }

    v8f acc;
    #pragma unroll
    for (int j = 0; j < 8; ++j) acc[j] = 0.0f;

    for (int c0 = 0; c0 < CCH; c0 += 32) {
        v8u af, bf;
        #pragma unroll
        for (int i = 0; i < 8; ++i) {
            int k = c0 + kofs(i, h);
            const float* wp = W + (size_t)(row0 + m) * CCH + k;   // A: row m, pair along K
            af[i] = packbf2(wp[0], wp[1]);
            bf[i] = *(const unsigned int*)(xT + ((size_t)b * NPIX + n0 + m) * CCH + k);
        }
        acc = wmma_bf16(af, bf, acc);
    }

    float vals[8];
    #pragma unroll
    for (int j = 0; j < 8; ++j) vals[j] = acc[j] + bias[row0 + j + 8 * h];

    if (mode == 2) {
        #pragma unroll
        for (int j = 0; j < 8; ++j)
            vb[((size_t)b * CCH + row0 + j + 8 * h) * NPIX + n0 + m] = f2bf(vals[j]);
    } else {
        unsigned short* dst = (mode == 0 ? qT : kT) +
            ((size_t)b * NPIX + n0 + m) * DQK + row0 + 8 * h;     // 16B aligned
        v4u pv = { packbf2(vals[0], vals[1]), packbf2(vals[2], vals[3]),
                   packbf2(vals[4], vals[5]), packbf2(vals[6], vals[7]) };
        *(v4u*)dst = pv;
    }
}

// ---------------------------------------------------------------------------
// 4) Flash position attention. One wave owns 16 queries x 128 channels
//    (chalf selects channel half). K loop in chunks of 32 keys:
//    S = Q(16x32) Kt  -> online softmax -> P through LDS (D->A transpose) ->
//    O += P(16x32) Vt. Epilogue: catT[b][q][c] = bf16(gamma*O/l + x).
// ---------------------------------------------------------------------------
__global__ __launch_bounds__(32) void k_pos_attn(
    const float* __restrict__ x,
    const unsigned short* __restrict__ qT, const unsigned short* __restrict__ kT,
    const unsigned short* __restrict__ vb,
    const float* __restrict__ gamma_pa,
    unsigned short* __restrict__ catT) {
    __shared__ unsigned short Pl[16 * 32];

    int wid   = blockIdx.x;                 // 4 * 256 * 2
    int b     = wid >> 9;
    int qtile = (wid >> 1) & 255;
    int c_base = (wid & 1) * 128;
    int q0 = qtile << 4;
    int L = threadIdx.x, h = L >> 4, m = L & 15;

    v8u qa;
    #pragma unroll
    for (int i = 0; i < 8; ++i)
        qa[i] = *(const unsigned int*)(qT + ((size_t)b * NPIX + q0 + m) * DQK + kofs(i, h));

    v8f oacc[8];
    float mr[8], lr[8];
    #pragma unroll
    for (int j = 0; j < 8; ++j) {
        mr[j] = -1e30f; lr[j] = 0.0f;
        #pragma unroll
        for (int t = 0; t < 8; ++t) oacc[t][j] = 0.0f;
    }

    for (int kk = 0; kk < NPIX; kk += 32) {
        v8u kb0, kb1;
        #pragma unroll
        for (int i = 0; i < 8; ++i) {
            int ko = kofs(i, h);
            kb0[i] = *(const unsigned int*)(kT + ((size_t)b * NPIX + kk + m)      * DQK + ko);
            kb1[i] = *(const unsigned int*)(kT + ((size_t)b * NPIX + kk + 16 + m) * DQK + ko);
        }
        // prefetch next key chunk while we do softmax math (global_prefetch_b8)
        if (kk + 32 < NPIX) {
            __builtin_prefetch(kT + ((size_t)b * NPIX + kk + 32 + m) * DQK, 0, 1);
            __builtin_prefetch(vb + ((size_t)b * CCH + c_base + m) * NPIX + kk + 32, 0, 1);
        }
        v8f z;
        #pragma unroll
        for (int j = 0; j < 8; ++j) z[j] = 0.0f;
        v8f s0 = wmma_bf16(qa, kb0, z);
        v8f s1 = wmma_bf16(qa, kb1, z);

        float scale[8], p0[8], p1[8];
        #pragma unroll
        for (int j = 0; j < 8; ++j) {
            float v = fmaxf(s0[j], s1[j]);                 // row max over 32 keys
            for (int off = 1; off < 16; off <<= 1) v = fmaxf(v, __shfl_xor(v, off, 16));
            float mn = fmaxf(mr[j], v);
            scale[j] = __expf(mr[j] - mn);
            p0[j] = __expf(s0[j] - mn);
            p1[j] = __expf(s1[j] - mn);
            float rs = p0[j] + p1[j];
            for (int off = 1; off < 16; off <<= 1) rs += __shfl_xor(rs, off, 16);
            lr[j] = lr[j] * scale[j] + rs;
            mr[j] = mn;
        }
        #pragma unroll
        for (int t = 0; t < 8; ++t)
            #pragma unroll
            for (int j = 0; j < 8; ++j) oacc[t][j] *= scale[j];

        // D-layout -> A-layout transpose of P through LDS (single wave owns it)
        #pragma unroll
        for (int j = 0; j < 8; ++j) {
            int row = j + 8 * h;
            Pl[row * 32 + m]      = f2bf(p0[j]);
            Pl[row * 32 + 16 + m] = f2bf(p1[j]);
        }
        asm volatile("s_wait_dscnt 0" ::: "memory");
        v8u pf;
        #pragma unroll
        for (int i = 0; i < 8; ++i)
            pf[i] = *(const unsigned int*)(Pl + m * 32 + kofs(i, h));

        #pragma unroll
        for (int t = 0; t < 8; ++t) {
            int c0 = c_base + 16 * t;
            v8u vf;
            #pragma unroll
            for (int i = 0; i < 8; ++i)
                vf[i] = *(const unsigned int*)(vb + ((size_t)b * CCH + c0 + m) * NPIX + kk + kofs(i, h));
            oacc[t] = wmma_bf16(pf, vf, oacc[t]);
        }
    }

    float g = gamma_pa[0];
    #pragma unroll
    for (int t = 0; t < 8; ++t) {
        int c = c_base + 16 * t + m;
        #pragma unroll
        for (int j = 0; j < 8; ++j) {
            int q = q0 + j + 8 * h;
            float o  = oacc[t][j] / lr[j];
            float xv = x[((size_t)b * CCH + c) * NPIX + q];
            catT[((size_t)b * NPIX + q) * 512 + c] = f2bf(g * o + xv);
        }
    }
}

// ---------------------------------------------------------------------------
// 5) Channel gram G = xb @ xb^T  [B,256,256] fp32 (WMMA, K = 4096)
// ---------------------------------------------------------------------------
__global__ __launch_bounds__(32) void k_gram(const unsigned short* __restrict__ xb,
                                             float* __restrict__ G) {
    int wid = blockIdx.x;                 // 4 * 256
    int b = wid >> 8;
    int t = wid & 255;
    int i0 = (t >> 4) << 4, j0 = (t & 15) << 4;
    int L = threadIdx.x, h = L >> 4, m = L & 15;

    v8f acc;
    #pragma unroll
    for (int j = 0; j < 8; ++j) acc[j] = 0.0f;

    for (int n = 0; n < NPIX; n += 32) {
        v8u af, bf;
        #pragma unroll
        for (int i = 0; i < 8; ++i) {
            int ko = n + kofs(i, h);
            af[i] = *(const unsigned int*)(xb + ((size_t)b * CCH + i0 + m) * NPIX + ko);
            bf[i] = *(const unsigned int*)(xb + ((size_t)b * CCH + j0 + m) * NPIX + ko);
        }
        acc = wmma_bf16(af, bf, acc);
    }
    #pragma unroll
    for (int j = 0; j < 8; ++j)
        G[((size_t)b * CCH + i0 + j + 8 * h) * CCH + j0 + m] = acc[j];
}

// ---------------------------------------------------------------------------
// 6) Row softmax of G -> cattnB bf16 [B,256,256]
// ---------------------------------------------------------------------------
__global__ __launch_bounds__(256) void k_chan_softmax(const float* __restrict__ G,
                                                      unsigned short* __restrict__ cattnB) {
    __shared__ float red[256];
    int b = blockIdx.x >> 8;
    int row = blockIdx.x & 255;
    int t = threadIdx.x;
    size_t base = ((size_t)b * CCH + row) * CCH;
    float v = G[base + t];
    red[t] = v; __syncthreads();
    for (int s = 128; s > 0; s >>= 1) { if (t < s) red[t] = fmaxf(red[t], red[t + s]); __syncthreads(); }
    float mx = red[0]; __syncthreads();
    float e = __expf(v - mx);
    red[t] = e; __syncthreads();
    for (int s = 128; s > 0; s >>= 1) { if (t < s) red[t] += red[t + s]; __syncthreads(); }
    cattnB[base + t] = f2bf(e / red[0]);
}

// ---------------------------------------------------------------------------
// 7) ca = cattn @ x, residual, write catT channels 256..511 (WMMA, K = 256)
// ---------------------------------------------------------------------------
__global__ __launch_bounds__(32) void k_ca_gemm(
    const float* __restrict__ x,
    const unsigned short* __restrict__ cattnB, const unsigned short* __restrict__ xT,
    const float* __restrict__ gamma_ca,
    unsigned short* __restrict__ catT) {
    int wid = blockIdx.x;                  // 4 * 16 * 256
    int b = wid / (16 * 256);
    int r = wid - b * 16 * 256;
    int i0 = (r >> 8) << 4;
    int n0 = (r & 255) << 4;
    int L = threadIdx.x, h = L >> 4, m = L & 15;

    v8f acc;
    #pragma unroll
    for (int j = 0; j < 8; ++j) acc[j] = 0.0f;

    for (int c0 = 0; c0 < CCH; c0 += 32) {
        v8u af, bf;
        #pragma unroll
        for (int i = 0; i < 8; ++i) {
            int k = c0 + kofs(i, h);
            af[i] = *(const unsigned int*)(cattnB + ((size_t)b * CCH + i0 + m) * CCH + k);
            bf[i] = *(const unsigned int*)(xT + ((size_t)b * NPIX + n0 + m) * CCH + k);
        }
        acc = wmma_bf16(af, bf, acc);
    }

    float g = gamma_ca[0];
    int n = n0 + m;
    float vals[8];
    #pragma unroll
    for (int j = 0; j < 8; ++j) {
        int ch = i0 + j + 8 * h;
        vals[j] = g * acc[j] + x[((size_t)b * CCH + ch) * NPIX + n];
    }
    unsigned short* dst = catT + ((size_t)b * NPIX + n) * 512 + 256 + i0 + 8 * h; // 16B aligned
    v4u pv = { packbf2(vals[0], vals[1]), packbf2(vals[2], vals[3]),
               packbf2(vals[4], vals[5]), packbf2(vals[6], vals[7]) };
    *(v4u*)dst = pv;
}

// ---------------------------------------------------------------------------
// 8) 3x3 conv (512->256) as 9 shifted implicit GEMMs over catT (K = 9*512).
// ---------------------------------------------------------------------------
__global__ __launch_bounds__(32) void k_conv(const unsigned short* __restrict__ wfr,
                                             const unsigned short* __restrict__ catT,
                                             float* __restrict__ y) {
    int wid = blockIdx.x;                  // 4 * 16 * 256
    int b = wid / (16 * 256);
    int r = wid - b * 16 * 256;
    int co0 = (r >> 8) << 4;
    int p0  = (r & 255) << 4;
    int L = threadIdx.x, h = L >> 4, m = L & 15;

    int p = p0 + m;                        // this lane's output pixel (B column)
    int py = p >> 6, px = p & 63;

    v8f acc;
    #pragma unroll
    for (int j = 0; j < 8; ++j) acc[j] = 0.0f;

    #pragma unroll
    for (int r9 = 0; r9 < 9; ++r9) {
        int dy = r9 / 3 - 1, dx = r9 % 3 - 1;
        int sy = py + dy, sx = px + dx;
        bool valid = ((unsigned)sy < 64u) && ((unsigned)sx < 64u);
        size_t srow = ((size_t)b * NPIX + (sy * 64 + sx)) * 512;
        for (int c0 = 0; c0 < 512; c0 += 32) {
            if (valid && c0 + 32 < 512)
                __builtin_prefetch(catT + srow + c0 + 32, 0, 1);
            v8u af, bf;
            #pragma unroll
            for (int i = 0; i < 8; ++i) {
                int k = c0 + kofs(i, h);
                af[i] = *(const unsigned int*)(wfr + ((size_t)r9 * 256 + co0 + m) * 512 + k);
                bf[i] = valid ? *(const unsigned int*)(catT + srow + k) : 0u;
            }
            acc = wmma_bf16(af, bf, acc);
        }
    }
    #pragma unroll
    for (int j = 0; j < 8; ++j)
        y[((size_t)b * CCH + co0 + j + 8 * h) * NPIX + p0 + m] = acc[j];
}

// ---------------------------------------------------------------------------
// 9) BN batch statistics per channel (mean, rsqrt(var+eps))
// ---------------------------------------------------------------------------
__global__ __launch_bounds__(256) void k_bn_stats(const float* __restrict__ y,
                                                  float* __restrict__ bnstats) {
    __shared__ float ssum[256];
    __shared__ float ssq[256];
    int c = blockIdx.x, t = threadIdx.x;
    float s = 0.0f, s2 = 0.0f;
    for (int i = t; i < BATCH * NPIX; i += 256) {
        int b = i >> 12, pix = i & (NPIX - 1);
        float v = y[((size_t)b * CCH + c) * NPIX + pix];
        s += v; s2 += v * v;
    }
    ssum[t] = s; ssq[t] = s2; __syncthreads();
    for (int st = 128; st > 0; st >>= 1) {
        if (t < st) { ssum[t] += ssum[t + st]; ssq[t] += ssq[t + st]; }
        __syncthreads();
    }
    if (t == 0) {
        float inv_n = 1.0f / (float)(BATCH * NPIX);
        float mean = ssum[0] * inv_n;
        float var  = ssq[0] * inv_n - mean * mean;     // biased, like torch BN
        bnstats[c]       = mean;
        bnstats[256 + c] = rsqrtf(var + 1e-5f);
    }
}

// ---------------------------------------------------------------------------
// 10) BN apply + ReLU -> d_out
// ---------------------------------------------------------------------------
__global__ __launch_bounds__(256) void k_bn_apply(const float* __restrict__ y,
                                                  const float* __restrict__ bnstats,
                                                  const float* __restrict__ bn_gamma,
                                                  const float* __restrict__ bn_beta,
                                                  float* __restrict__ out) {
    int idx = blockIdx.x * 256 + threadIdx.x;       // 4*256*4096
    int c = (idx >> 12) & 255;
    float v = (y[idx] - bnstats[c]) * bnstats[256 + c];
    out[idx] = fmaxf(0.0f, bn_gamma[c] * v + bn_beta[c]);
}

// ---------------------------------------------------------------------------
extern "C" void kernel_launch(void* const* d_in, const int* in_sizes, int n_in,
                              void* d_out, int out_size, void* d_ws, size_t ws_size,
                              hipStream_t stream) {
    (void)in_sizes; (void)n_in; (void)out_size; (void)ws_size;
    const float* x        = (const float*)d_in[0];
    const float* wq       = (const float*)d_in[1];
    const float* bq       = (const float*)d_in[2];
    const float* wk       = (const float*)d_in[3];
    const float* bk       = (const float*)d_in[4];
    const float* wv       = (const float*)d_in[5];
    const float* bv       = (const float*)d_in[6];
    const float* gamma_pa = (const float*)d_in[7];
    const float* gamma_ca = (const float*)d_in[8];
    const float* wf       = (const float*)d_in[9];
    const float* bn_gamma = (const float*)d_in[10];
    const float* bn_beta  = (const float*)d_in[11];
    float* out = (float*)d_out;

    // workspace carve-out (all sizes 256B aligned)
    char* base = (char*)d_ws;
    size_t off = 0;
    auto carve = [&](size_t bytes) -> char* {
        char* p = base + off;
        off = (off + bytes + 255) & ~(size_t)255;
        return p;
    };
    unsigned short* xb     = (unsigned short*)carve((size_t)BATCH * CCH * NPIX * 2);  // 8 MB
    unsigned short* xT     = (unsigned short*)carve((size_t)BATCH * NPIX * CCH * 2);  // 8 MB
    unsigned short* qT     = (unsigned short*)carve((size_t)BATCH * NPIX * DQK * 2);  // 1 MB
    unsigned short* kT     = (unsigned short*)carve((size_t)BATCH * NPIX * DQK * 2);  // 1 MB
    unsigned short* vb     = (unsigned short*)carve((size_t)BATCH * CCH * NPIX * 2);  // 8 MB
    float*          G      = (float*)carve((size_t)BATCH * CCH * CCH * 4);            // 1 MB
    unsigned short* cattnB = (unsigned short*)carve((size_t)BATCH * CCH * CCH * 2);   // 0.5 MB
    unsigned short* catT   = (unsigned short*)carve((size_t)BATCH * NPIX * 512 * 2);  // 16 MB
    unsigned short* wfr    = (unsigned short*)carve((size_t)9 * 256 * 512 * 2);       // 2.25 MB
    float*          y      = (float*)carve((size_t)BATCH * CCH * NPIX * 4);           // 16 MB
    float*          bnst   = (float*)carve(512 * 4);

    const int NTOT = BATCH * CCH * NPIX;   // 4,194,304

    k_cvt_x<<<NTOT / 256, 256, 0, stream>>>(x, xb, xT);
    k_cvt_w<<<(256 * 512 * 9 + 255) / 256, 256, 0, stream>>>(wf, wfr);
    k_proj<<<BATCH * 20 * 256, 32, 0, stream>>>(wq, bq, wk, bk, wv, bv, xT, qT, kT, vb);
    k_pos_attn<<<BATCH * 256 * 2, 32, 0, stream>>>(x, qT, kT, vb, gamma_pa, catT);
    k_gram<<<BATCH * 256, 32, 0, stream>>>(xb, G);
    k_chan_softmax<<<BATCH * 256, 256, 0, stream>>>(G, cattnB);
    k_ca_gemm<<<BATCH * 16 * 256, 32, 0, stream>>>(x, cattnB, xT, gamma_ca, catT);
    k_conv<<<BATCH * 16 * 256, 32, 0, stream>>>(wfr, catT, y);
    k_bn_stats<<<256, 256, 0, stream>>>(y, bnst);
    k_bn_apply<<<NTOT / 256, 256, 0, stream>>>(y, bnst, bn_gamma, bn_beta, out);
}
